// MEGAFeatureExtractor_61701500175293
// MI455X (gfx1250) — matrix operand order repaired and checked by reading, not compiled, not run on hardware
//
#include <hip/hip_runtime.h>

// ---------------- constants (match reference) ----------------
#define NPTS   6000
#define DDIM   1024
#define GRP    16
#define DG     64
#define KSEL   300
#define EMBD   64

typedef unsigned short u16;
typedef __attribute__((ext_vector_type(16))) __bf16 v16bf;
typedef __attribute__((ext_vector_type(8)))  float  v8f;
typedef __attribute__((ext_vector_type(4)))  unsigned int vu4;
typedef __attribute__((ext_vector_type(8)))  int  vi8;
typedef __attribute__((ext_vector_type(4)))  int  vi4;

union FragBF { vu4 u[2]; v16bf v; };

#if __has_builtin(__builtin_amdgcn_tensor_load_to_lds)
#define TDM_AVAILABLE 1
#else
#define TDM_AVAILABLE 0
#endif

// ---------------- helpers ----------------
__global__ void sqnorm_kernel(const float* __restrict__ feat, float* __restrict__ sqn) {
    int j = blockIdx.x * blockDim.x + threadIdx.x;
    if (j >= NPTS) return;
    const float4* x = (const float4*)(feat + (size_t)j * DDIM);
    float s = 0.f;
    for (int i = 0; i < DDIM / 4; ++i) {
        float4 a = x[i];
        s += a.x * a.x + a.y * a.y + a.z * a.z + a.w * a.w;
    }
    sqn[j] = s;
}

__global__ void to_bf16_kernel(const float* __restrict__ src, u16* __restrict__ dst, int n) {
    int i = blockIdx.x * blockDim.x + threadIdx.x;
    if (i >= n) return;
    unsigned int x = __float_as_uint(src[i]);
    unsigned int r = x + 0x7FFFu + ((x >> 16) & 1u);   // round-to-nearest-even bf16
    dst[i] = (u16)(r >> 16);
}

// ---------------- FPS greedy (serial over steps, parallel within a step) ----------------
__global__ void fps_init_kernel(int* idxList) {
    if (threadIdx.x == 0) idxList[0] = 0;
}

__global__ __launch_bounds__(128) void fps_step_kernel(
        const float* __restrict__ feat, const float* __restrict__ sqn,
        float* __restrict__ ds, const int* __restrict__ idxList, int t,
        float* __restrict__ pVal, int* __restrict__ pIdx) {
    const int cur = idxList[t - 1];
    const int j = blockIdx.x * blockDim.x + threadIdx.x;
    float val = __builtin_inff();
    int   vidx = 0x7fffffff;
    if (j < NPTS) {
        const float4* xj = (const float4*)(feat + (size_t)j   * DDIM);
        const float4* xc = (const float4*)(feat + (size_t)cur * DDIM);
        float dot = 0.f;
        for (int i = 0; i < DDIM / 4; ++i) {
            float4 a = xj[i], b = xc[i];
            dot += a.x * b.x + a.y * b.y + a.z * b.z + a.w * b.w;
        }
        float sq = sqn[cur] + sqn[j] - 2.f * dot;
        float r = (j == cur || !(sq > 0.f)) ? __builtin_inff() : rsqrtf(sq); // 1/cdist, diag=inf
        float d = ((t == 1) ? 0.f : ds[j]) + r;   // ds starts as Drec[0] at t==1
        ds[j] = d;
        val = d; vidx = j;
    }
    __shared__ float sv[128];
    __shared__ int   si[128];
    sv[threadIdx.x] = val; si[threadIdx.x] = vidx;
    __syncthreads();
    for (int s = 64; s > 0; s >>= 1) {
        if (threadIdx.x < s) {
            float ov = sv[threadIdx.x + s]; int oi = si[threadIdx.x + s];
            if (ov < sv[threadIdx.x] || (ov == sv[threadIdx.x] && oi < si[threadIdx.x])) {
                sv[threadIdx.x] = ov; si[threadIdx.x] = oi;
            }
        }
        __syncthreads();
    }
    if (threadIdx.x == 0) { pVal[blockIdx.x] = sv[0]; pIdx[blockIdx.x] = si[0]; }
}

__global__ void fps_reduce_kernel(const float* __restrict__ pVal, const int* __restrict__ pIdx,
                                  int* __restrict__ idxList, int t, int nparts) {
    __shared__ float sv[64];
    __shared__ int   si[64];
    int i = threadIdx.x;
    sv[i] = (i < nparts) ? pVal[i] : __builtin_inff();
    si[i] = (i < nparts) ? pIdx[i] : 0x7fffffff;
    __syncthreads();
    for (int s = 32; s > 0; s >>= 1) {
        if (i < s) {
            float ov = sv[i + s]; int oi = si[i + s];
            if (ov < sv[i] || (ov == sv[i] && oi < si[i])) { sv[i] = ov; si[i] = oi; }
        }
        __syncthreads();
    }
    if (i == 0) idxList[t] = si[0];
}

__global__ void gather_kernel(const u16* __restrict__ featBf, const float* __restrict__ bboxes,
                              const int* __restrict__ idxList,
                              u16* __restrict__ roiBf, float* __restrict__ roiBox) {
    int k = blockIdx.x;
    int src = idxList[k];
    for (int i = threadIdx.x; i < DDIM; i += blockDim.x)
        roiBf[(size_t)k * DDIM + i] = featBf[(size_t)src * DDIM + i];
    if (threadIdx.x < 4)
        roiBox[k * 4 + threadIdx.x] = bboxes[src * 4 + threadIdx.x];
}

// ---------------- TDM tile stage: 64 rows x 32 bf16, row stride 1024 elements ----------------
// D# packed per CDNA5 ISA. 2D tensor -> groups 2/3 zero.
__device__ __forceinline__ void tdm_load_tile_bf16(unsigned ldsOff, const u16* gptr,
                                                   int remRows, int remCols) {
#if TDM_AVAILABLE
    unsigned long long ga = (unsigned long long)(size_t)gptr;
    vu4 g0;
    g0[0] = 1u;                                        // count=1, user mode
    g0[1] = ldsOff;                                    // lds_addr [63:32]
    g0[2] = (unsigned)(ga & 0xffffffffull);            // global_addr [95:64]
    g0[3] = (unsigned)((ga >> 32) & 0x01ffffffull)     // global_addr [120:96]
            | (2u << 30);                              // type=2 ("image") [127:126]
    unsigned td0 = (unsigned)remCols;                  // tensor extent dim0 (elements)
    unsigned td1 = (unsigned)remRows;                  // tensor extent dim1 (rows)
    vi8 g1;
    g1[0] = (int)(1u << 16);                           // workgroup_mask=0, data_size=1 (2B)
    g1[1] = (int)((td0 & 0xffffu) << 16);              // tensor_dim0[15:0] @ bits 63:48
    g1[2] = (int)(((td0 >> 16) & 0xffffu)              // tensor_dim0[31:16]
            | ((td1 & 0xffffu) << 16));                // tensor_dim1[15:0]
    g1[3] = (int)(((td1 >> 16) & 0xffffu)              // tensor_dim1[31:16]
            | (32u << 16));                            // tile_dim0 = 32
    g1[4] = (int)64u;                                  // tile_dim1 = 64, tile_dim2 = 0
    g1[5] = (int)DDIM;                                 // tensor_dim0_stride = 1024
    g1[6] = 0;
    g1[7] = 0;
    vi4 z4 = {0, 0, 0, 0};
#if __clang_major__ >= 23
    vi8 z8 = {0, 0, 0, 0, 0, 0, 0, 0};
    __builtin_amdgcn_tensor_load_to_lds(g0, g1, z4, z4, z8, 0);
#else
    __builtin_amdgcn_tensor_load_to_lds(g0, g1, z4, z4, 0);
#endif
#else
    (void)ldsOff; (void)gptr; (void)remRows; (void)remCols;
#endif
}

// ---------------- WMMA bf16 GEMM: C[M,1024] = A[M,1024] * W[1024,1024]^T + bias --------------
// Block = 128 threads = 4 waves, computes a 64x64 tile. Each wave: 32x32 (2x2 WMMA frags).
// A/B K-slices (64x32 bf16) staged to LDS by the Tensor Data Mover, double-buffered,
// synced with s_wait_tensorcnt + workgroup barriers. Fragments read from LDS (ds_load_b128).
__global__ __launch_bounds__(128) void gemm_bf16_kernel(
        const u16* __restrict__ A, const u16* __restrict__ W,
        const float* __restrict__ bias, float* __restrict__ C, int M) {
    __shared__ u16 ldsA[2][64 * 32];
    __shared__ u16 ldsB[2][64 * 32];

    const int tid  = threadIdx.x;
    const int lane = tid & 31;
    const int wave = tid >> 5;
    const int m0 = blockIdx.x * 64;
    const int n0 = blockIdx.y * 64;
    const int wm = (wave >> 1) * 32;     // wave's M offset inside block tile
    const int wn = (wave & 1) * 32;      // wave's N offset inside block tile
    const int hi  = lane >> 4;
    const int l16 = lane & 15;
    const int kbA = hi * 8;              // A frag: K runs {kbA..+7} and {kbA+16..+23}
    const int kbB = hi * 8;              // B frag: K runs {kbB..+7} and {kbB+16..+23}

    const u16* Abase = A + (size_t)m0 * DDIM;
    const u16* Wbase = W + (size_t)n0 * DDIM;
    const int remA = M - m0;             // A rows remaining at tile start (OOB rows -> zero)

#if TDM_AVAILABLE
    unsigned offA[2] = { (unsigned)(size_t)&ldsA[0][0], (unsigned)(size_t)&ldsA[1][0] };
    unsigned offB[2] = { (unsigned)(size_t)&ldsB[0][0], (unsigned)(size_t)&ldsB[1][0] };
    if (wave == 0) {                     // prologue: stage k0 = 0 into buffer 0
        tdm_load_tile_bf16(offA[0], Abase, remA, DDIM);
        tdm_load_tile_bf16(offB[0], Wbase, 1024 - n0, DDIM);
    }
#endif

    v8f acc00 = {0.f,0.f,0.f,0.f,0.f,0.f,0.f,0.f};
    v8f acc01 = acc00, acc10 = acc00, acc11 = acc00;

    for (int k0 = 0, it = 0; k0 < DDIM; k0 += 32, ++it) {
        const int cur = it & 1;
        __syncthreads();                 // all waves done reading buffer[cur^1]
#if TDM_AVAILABLE
        if (wave == 0) {
            if ((k0 + 32) < DDIM) {
                tdm_load_tile_bf16(offA[cur ^ 1], Abase + (k0 + 32), remA, DDIM - (k0 + 32));
                tdm_load_tile_bf16(offB[cur ^ 1], Wbase + (k0 + 32), 1024 - n0, DDIM - (k0 + 32));
                // in-order TENSORcnt: <=2 leaves only the just-issued pair outstanding
                __builtin_amdgcn_s_wait_tensorcnt(2);
            } else {
                __builtin_amdgcn_s_wait_tensorcnt(0);
            }
        }
#else
        for (int c = tid; c < 256; c += 128) {       // fallback: cooperative copy A tile
            int row = c >> 2, part = c & 3;
            vu4 v = {0, 0, 0, 0};
            if (row < remA) v = *(const vu4*)(Abase + (size_t)row * DDIM + k0 + part * 8);
            *(vu4*)(&ldsA[cur][0] + row * 32 + part * 8) = v;
        }
        for (int c = tid; c < 256; c += 128) {       // B tile (always in-bounds)
            int row = c >> 2, part = c & 3;
            *(vu4*)(&ldsB[cur][0] + row * 32 + part * 8) =
                *(const vu4*)(Wbase + (size_t)row * DDIM + k0 + part * 8);
        }
#endif
        __syncthreads();                 // buffer[cur] ready for everyone

        const u16* la = &ldsA[cur][0];
        const u16* lb = &ldsB[cur][0];
        FragBF a0, a1, b0, b1;
        a0.u[0] = *(const vu4*)(la + (wm      + l16) * 32 + kbA);
        a0.u[1] = *(const vu4*)(la + (wm      + l16) * 32 + kbA + 16);
        a1.u[0] = *(const vu4*)(la + (wm + 16 + l16) * 32 + kbA);
        a1.u[1] = *(const vu4*)(la + (wm + 16 + l16) * 32 + kbA + 16);
        b0.u[0] = *(const vu4*)(lb + (wn      + l16) * 32 + kbB);
        b0.u[1] = *(const vu4*)(lb + (wn      + l16) * 32 + kbB + 16);
        b1.u[0] = *(const vu4*)(lb + (wn + 16 + l16) * 32 + kbB);
        b1.u[1] = *(const vu4*)(lb + (wn + 16 + l16) * 32 + kbB + 16);

        acc00 = __builtin_amdgcn_wmma_f32_16x16x32_bf16(false, a0.v, false, b0.v, (short)0, acc00, false, false);
        acc01 = __builtin_amdgcn_wmma_f32_16x16x32_bf16(false, a0.v, false, b1.v, (short)0, acc01, false, false);
        acc10 = __builtin_amdgcn_wmma_f32_16x16x32_bf16(false, a1.v, false, b0.v, (short)0, acc10, false, false);
        acc11 = __builtin_amdgcn_wmma_f32_16x16x32_bf16(false, a1.v, false, b1.v, (short)0, acc11, false, false);
    }

    const int nA = n0 + wn + l16;
    const int nB = n0 + wn + 16 + l16;
    const float bn0 = bias ? bias[nA] : 0.f;
    const float bn1 = bias ? bias[nB] : 0.f;
    for (int r = 0; r < 8; ++r) {
        int mA = m0 + wm + r + hi * 8;           // D layout: vgpr r -> row r (+8 for hi lanes)
        int mB = mA + 16;
        if (mA < M) {
            C[(size_t)mA * DDIM + nA] = acc00[r] + bn0;
            C[(size_t)mA * DDIM + nB] = acc01[r] + bn1;
        }
        if (mB < M) {
            C[(size_t)mB * DDIM + nA] = acc10[r] + bn0;
            C[(size_t)mB * DDIM + nB] = acc11[r] + bn1;
        }
    }
}

// ---------------- fused position-gate + multi-head attention (flash-style, 1 block per query) ----
__global__ __launch_bounds__(256) void attn_kernel(
        const float* __restrict__ Qp, const float* __restrict__ Kp, const float* __restrict__ Vp,
        const float* __restrict__ bboxes, const float* __restrict__ roiBox,
        const float* __restrict__ Wg, const float* __restrict__ bg,
        const float* __restrict__ bv, float* __restrict__ out) {
    const int k = blockIdx.x;
    const int tid = threadIdx.x;
    __shared__ float qs [GRP][DG];
    __shared__ float wgs[GRP][EMBD];
    __shared__ float bgs[GRP];
    __shared__ float mulf[8];
    __shared__ float embt[16][EMBD];
    __shared__ float logit[GRP][17];
    __shared__ float boxT[16][4];
    __shared__ float roiG[4];

    for (int i = tid; i < DDIM; i += 256) {
        qs [i >> 6][i & 63] = Qp[(size_t)k * DDIM + i];
        wgs[i >> 6][i & 63] = Wg[i];
    }
    if (tid < GRP) bgs[tid] = bg[tid];
    if (tid < 8)   mulf[tid] = 100.0f / powf(1000.0f, (float)tid * 0.125f);
    if (tid == 0) {
        float x1 = roiBox[k*4+0], y1 = roiBox[k*4+1], x2 = roiBox[k*4+2], y2 = roiBox[k*4+3];
        roiG[0] = x2 - x1 + 1.f; roiG[1] = y2 - y1 + 1.f;
        roiG[2] = 0.5f * (x1 + x2); roiG[3] = 0.5f * (y1 + y2);
    }
    __syncthreads();

    const int g = tid >> 4;
    const int l16 = tid & 15;
    float4 acc = {0.f, 0.f, 0.f, 0.f};
    float mrun = -__builtin_inff();
    float lrun = 0.f;

    for (int n0 = 0; n0 < NPTS; n0 += 16) {
        if (tid < 64) boxT[tid >> 2][tid & 3] = bboxes[(n0 + (tid >> 2)) * 4 + (tid & 3)];
        __syncthreads();

        {   // embeddings: thread -> (n_local = tid>>4, 4 features from (tid&15)*4)
            int nl = tid >> 4;
            int e0 = (tid & 15) * 4;
            float x1 = boxT[nl][0], y1 = boxT[nl][1], x2 = boxT[nl][2], y2 = boxT[nl][3];
            float wr = x2 - x1 + 1.f, hr = y2 - y1 + 1.f;
            float cxr = 0.5f * (x1 + x2), cyr = 0.5f * (y1 + y2);
            float wk_ = roiG[0], hk = roiG[1], cxk = roiG[2], cyk = roiG[3];
            float pos[4];
            pos[0] = logf(fabsf((cxk - cxr) / wk_) + 1e-3f);
            pos[1] = logf(fabsf((cyk - cyr) / hk ) + 1e-3f);
            pos[2] = logf(wk_ / wr);
            pos[3] = logf(hk  / hr);
            for (int e = e0; e < e0 + 4; ++e) {
                int c = e >> 4, rem = e & 15, f = rem & 7;
                float v = pos[c] * mulf[f];
                embt[nl][e] = (rem & 8) ? cosf(v) : sinf(v);
            }
        }
        __syncthreads();

        {   // logits: thread -> (g, n_local = l16)
            float gate = bgs[g];
            for (int e = 0; e < EMBD; ++e) gate += wgs[g][e] * embt[l16][e];
            gate = fmaxf(gate, 0.f);
            float aff = 0.f;
            const float4* kp = (const float4*)(Kp + (size_t)(n0 + l16) * DDIM + g * DG);
            if (n0 + 16 < NPTS)
                __builtin_prefetch(Kp + (size_t)(n0 + 16 + l16) * DDIM + g * DG, 0, 0);
            const float* qg = qs[g];
            for (int d = 0; d < 16; ++d) {
                float4 kv = kp[d];
                aff += qg[d*4+0]*kv.x + qg[d*4+1]*kv.y + qg[d*4+2]*kv.z + qg[d*4+3]*kv.w;
            }
            logit[g][l16] = logf(gate + 1e-6f) + aff * 0.125f;
        }
        __syncthreads();

        {   // online softmax accumulate: thread -> (g, 4 out dims at l16*4)
            float tmax = -__builtin_inff();
            for (int nl = 0; nl < 16; ++nl) tmax = fmaxf(tmax, logit[g][nl]);
            float mnew = fmaxf(mrun, tmax);
            float scale = __expf(mrun - mnew);
            acc.x *= scale; acc.y *= scale; acc.z *= scale; acc.w *= scale;
            lrun *= scale;
            const float4* vp = (const float4*)Vp;
            for (int nl = 0; nl < 16; ++nl) {
                float p = __expf(logit[g][nl] - mnew);
                float4 v = vp[(size_t)(n0 + nl) * (DDIM / 4) + g * 16 + l16];
                acc.x += p * v.x; acc.y += p * v.y; acc.z += p * v.z; acc.w += p * v.w;
                lrun += p;
            }
            mrun = mnew;
        }
        __syncthreads();
    }

    float inv = 1.f / lrun;
    int ob = k * DDIM + g * DG + l16 * 4;
    out[ob + 0] = acc.x * inv + bv[g * DG + l16 * 4 + 0];
    out[ob + 1] = acc.y * inv + bv[g * DG + l16 * 4 + 1];
    out[ob + 2] = acc.z * inv + bv[g * DG + l16 * 4 + 2];
    out[ob + 3] = acc.w * inv + bv[g * DG + l16 * 4 + 3];
}

// ---------------- host orchestration ----------------
extern "C" void kernel_launch(void* const* d_in, const int* in_sizes, int n_in,
                              void* d_out, int out_size, void* d_ws, size_t ws_size,
                              hipStream_t stream) {
    const float* feat   = (const float*)d_in[0];
    const float* bboxes = (const float*)d_in[1];
    const float* Wq     = (const float*)d_in[2];
    const float* bq     = (const float*)d_in[3];
    const float* Wk     = (const float*)d_in[4];
    const float* bk     = (const float*)d_in[5];
    const float* Wg     = (const float*)d_in[6];
    const float* bg     = (const float*)d_in[7];
    const float* Wv     = (const float*)d_in[8];   // (16,64,1024) == row-major (1024,1024)
    const float* bv     = (const float*)d_in[9];
    float* out = (float*)d_out;

    char* ws = (char*)d_ws;
    size_t off = 0;
    auto alloc = [&](size_t bytes) -> char* {
        char* p = ws + off;
        off = (off + bytes + 255) & ~(size_t)255;
        return p;
    };
    float* sqn     = (float*)alloc(NPTS * sizeof(float));
    float* ds      = (float*)alloc(NPTS * sizeof(float));
    int*   idxList = (int*)  alloc(KSEL * sizeof(int));
    float* pVal    = (float*)alloc(64 * sizeof(float));
    int*   pIdx    = (int*)  alloc(64 * sizeof(int));
    u16*   featBf  = (u16*)  alloc((size_t)NPTS * DDIM * sizeof(u16));
    u16*   WqBf    = (u16*)  alloc((size_t)DDIM * DDIM * sizeof(u16));
    u16*   WkBf    = (u16*)  alloc((size_t)DDIM * DDIM * sizeof(u16));
    u16*   WvBf    = (u16*)  alloc((size_t)DDIM * DDIM * sizeof(u16));
    u16*   roiBf   = (u16*)  alloc((size_t)KSEL * DDIM * sizeof(u16));
    float* roiBox  = (float*)alloc(KSEL * 4 * sizeof(float));
    float* Qp      = (float*)alloc((size_t)KSEL * DDIM * sizeof(float));
    float* Kp      = (float*)alloc((size_t)NPTS * DDIM * sizeof(float));
    float* Vp      = (float*)alloc((size_t)NPTS * DDIM * sizeof(float));
    (void)ws_size; (void)in_sizes; (void)n_in; (void)out_size;

    // 1) row squared-norms + bf16 copies (everything stays L2-resident, ~70 MB < 192 MB L2)
    sqnorm_kernel<<<(NPTS + 127) / 128, 128, 0, stream>>>(feat, sqn);
    {
        int nf = NPTS * DDIM;
        to_bf16_kernel<<<(nf + 255) / 256, 256, 0, stream>>>(feat, featBf, nf);
        int nw = DDIM * DDIM;
        to_bf16_kernel<<<(nw + 255) / 256, 256, 0, stream>>>(Wq, WqBf, nw);
        to_bf16_kernel<<<(nw + 255) / 256, 256, 0, stream>>>(Wk, WkBf, nw);
        to_bf16_kernel<<<(nw + 255) / 256, 256, 0, stream>>>(Wv, WvBf, nw);
    }

    // 2) greedy FPS: 299 dependent steps, each parallel across the device
    fps_init_kernel<<<1, 32, 0, stream>>>(idxList);
    const int STEP_BLOCKS = (NPTS + 127) / 128;   // 47
    for (int t = 1; t < KSEL; ++t) {
        fps_step_kernel<<<STEP_BLOCKS, 128, 0, stream>>>(feat, sqn, ds, idxList, t, pVal, pIdx);
        fps_reduce_kernel<<<1, 64, 0, stream>>>(pVal, pIdx, idxList, t, STEP_BLOCKS);
    }

    // 3) gather selected rows
    gather_kernel<<<KSEL, 256, 0, stream>>>(featBf, bboxes, idxList, roiBf, roiBox);

    // 4) WMMA projections (TDM-staged, double-buffered LDS tiles):
    //    Qp = roi*Wq^T+bq, Kp = feat*Wk^T+bk, Vp = feat*Wv^T (bv folded into final output)
    gemm_bf16_kernel<<<dim3((KSEL + 63) / 64, DDIM / 64), 128, 0, stream>>>(roiBf, WqBf, bq, Qp, KSEL);
    gemm_bf16_kernel<<<dim3((NPTS + 63) / 64, DDIM / 64), 128, 0, stream>>>(featBf, WkBf, bk, Kp, NPTS);
    gemm_bf16_kernel<<<dim3((NPTS + 63) / 64, DDIM / 64), 128, 0, stream>>>(featBf, WvBf, (const float*)nullptr, Vp, NPTS);

    // 5) fused position-gate + attention, one block per query
    attn_kernel<<<KSEL, 256, 0, stream>>>(Qp, Kp, Vp, bboxes, roiBox, Wg, bg, bv, out);
}
